// TransformerDecoderBlock_41901700939833
// MI455X (gfx1250) — matrix operand order repaired and checked by reading, not compiled, not run on hardware
//
#include <hip/hip_runtime.h>

typedef _Float16 h16;
typedef __attribute__((ext_vector_type(16))) _Float16 v16h;
typedef __attribute__((ext_vector_type(8)))  _Float16 v8h;
typedef __attribute__((ext_vector_type(8)))  float    v8f;

#define V_SZ 50257
#define T_SZ 2048
#define E_SZ 768
#define H_SZ 12
#define DH   64
#define B_SZ 2

// ---------- WMMA helpers (CDNA5 16x16x32 f16 -> f32) ----------

__device__ __forceinline__ v8f wmma_f16(v16h a, v16h b, v8f c) {
  // 8 args: (neg_a, A, neg_b, B, c_mod, C, reuse_a, reuse_b)
  return __builtin_amdgcn_wmma_f32_16x16x32_f16(false, a, false, b, (short)0, c,
                                                false, false);
}

// A fragment: 16x32 f16, row-major source with leading dim `ld`.
// lane<16: row=lane,   elems 0..15 = K 0..15
// lane>=16: row=lane-16, elems 0..15 = K 16..31
__device__ __forceinline__ v16h frag_a(const h16* base, int row0, int k0, int ld,
                                       int lane) {
  const h16* p = base + (size_t)(row0 + (lane & 15)) * ld + k0 + ((lane >> 4) << 4);
  v8h lo = *(const v8h*)p;
  v8h hi = *(const v8h*)(p + 8);
  return __builtin_shufflevector(lo, hi, 0, 1, 2, 3, 4, 5, 6, 7, 8, 9, 10, 11, 12,
                                 13, 14, 15);
}

// B fragment: 32x16 f16 taken from B-transposed storage BT[N][K] (row-major, ld).
// lane<16: col=lane, K 0..15 contiguous; lane>=16: col=lane-16, K 16..31.
__device__ __forceinline__ v16h frag_b(const h16* bt, int n0, int k0, int ld,
                                       int lane) {
  const h16* p = bt + (size_t)(n0 + (lane & 15)) * ld + k0 + ((lane >> 4) << 4);
  return *(const v16h*)p;
}

// ---------- CDNA5 async global->LDS copy (ASYNCcnt path) ----------
__device__ __forceinline__ void async_copy_b128(unsigned lds_off, const void* gptr) {
  asm volatile("global_load_async_to_lds_b128 %0, %1, off"
               :
               : "v"(lds_off), "v"(gptr)
               : "memory");
}
__device__ __forceinline__ void wait_async0() {
  asm volatile("s_wait_asynccnt 0" ::: "memory");
}

// ---------- 1) embedding ----------
__global__ void embed_kernel(const int* x, const float* tok, const float* pos,
                             h16* hbuf) {
  int i = blockIdx.x * 256 + threadIdx.x;
  if (i >= B_SZ * T_SZ * E_SZ) return;
  int e = i % E_SZ;
  int bt = i / E_SZ;
  int t = bt % T_SZ;
  float v = tok[(size_t)x[bt] * E_SZ + e] + pos[(size_t)t * E_SZ + e];
  hbuf[i] = (h16)v;
}

// ---------- 2) weight prep ----------
__global__ void prep_w_kernel(const float* Wq, const float* Wk, const float* Wv,
                              h16* Wqt, h16* Wkt, h16* Wvt) {
  int i = blockIdx.x * 256 + threadIdx.x;
  const int N = H_SZ * E_SZ * DH;
  if (i >= N) return;
  int hh = i / (E_SZ * DH);
  int rem = i % (E_SZ * DH);
  int e = rem / DH;
  int d = rem % DH;
  int o = hh * DH * E_SZ + d * E_SZ + e;  // [H, Dh, E]
  Wqt[o] = (h16)Wq[i];
  Wkt[o] = (h16)Wk[i];
  Wvt[o] = (h16)Wv[i];
}

__global__ void prep_wout_kernel(const float* Wout, h16* WoutT) {
  size_t i = (size_t)blockIdx.x * 256 + threadIdx.x;
  if (i >= (size_t)E_SZ * V_SZ) return;
  int e = (int)(i / V_SZ);
  int v = (int)(i % V_SZ);
  WoutT[(size_t)v * E_SZ + e] = (h16)Wout[i];  // [V, E]
}

// ---------- 3) QKV projections ----------
// q,k: [B*H, T, 64]   vT: [B*H, 64, T]
__global__ void qkv_kernel(const h16* hbuf, const h16* Wqt, const h16* Wkt,
                           const h16* Wvt, h16* q, h16* k, h16* vT) {
  int lane = threadIdx.x & 31;
  int wave = threadIdx.x >> 5;
  int bh = blockIdx.y;
  int b = bh / H_SZ;
  int head = bh % H_SZ;
  int t0 = (blockIdx.x * 8 + wave) * 16;
  int rowA = b * T_SZ + t0;
  const h16* Ws[3] = {Wqt + (size_t)head * DH * E_SZ, Wkt + (size_t)head * DH * E_SZ,
                      Wvt + (size_t)head * DH * E_SZ};
  int mrow = ((lane >> 4) << 3);
  for (int p = 0; p < 3; ++p) {
    v8f acc[4] = {{}, {}, {}, {}};
    for (int k0 = 0; k0 < E_SZ; k0 += 32) {
      v16h a = frag_a(hbuf, rowA, k0, E_SZ, lane);
#pragma unroll
      for (int nt = 0; nt < 4; ++nt)
        acc[nt] = wmma_f16(a, frag_b(Ws[p], nt * 16, k0, E_SZ, lane), acc[nt]);
    }
    if (p < 2) {
      h16* dst = (p == 0 ? q : k) + ((size_t)bh * T_SZ + t0) * DH;
#pragma unroll
      for (int nt = 0; nt < 4; ++nt)
        for (int r = 0; r < 8; ++r)
          dst[(size_t)(mrow + r) * DH + nt * 16 + (lane & 15)] = (h16)acc[nt][r];
    } else {
      h16* dst = vT + (size_t)bh * DH * T_SZ;
#pragma unroll
      for (int nt = 0; nt < 4; ++nt) {
        int d = nt * 16 + (lane & 15);
        for (int r = 0; r < 8; ++r)
          dst[(size_t)d * T_SZ + t0 + mrow + r] = (h16)acc[nt][r];
      }
    }
  }
}

// ---------- 4) per-key-column online max / sum over the QUERY axis ----------
__global__ void colstats_kernel(const h16* q, const h16* k, float* cm, float* cs) {
  int lane = threadIdx.x & 31;
  int wave = threadIdx.x >> 5;
  int bh = blockIdx.y;
  int s0 = (blockIdx.x * 4 + wave) * 16;
  const h16* qb = q + (size_t)bh * T_SZ * DH;
  const h16* kb = k + (size_t)bh * T_SZ * DH;
  v16h b0 = frag_b(kb, s0, 0, DH, lane);
  v16h b1 = frag_b(kb, s0, 32, DH, lane);
  int s = s0 + (lane & 15);
  int mrow = ((lane >> 4) << 3);
  float m_run = -1.0e30f, s_run = 0.0f;
  for (int t0 = s0; t0 < T_SZ; t0 += 16) {
    v8f c = {};
    c = wmma_f16(frag_a(qb, t0, 0, DH, lane), b0, c);
    c = wmma_f16(frag_a(qb, t0, 32, DH, lane), b1, c);
    float vals[8];
    float tmax = -3.0e38f;
#pragma unroll
    for (int r = 0; r < 8; ++r) {
      int t = t0 + mrow + r;
      vals[r] = (t >= s) ? 8.0f * c[r] : -3.0e38f;  // * sqrt(Dh), causal mask
      tmax = fmaxf(tmax, vals[r]);
    }
    float m_new = fmaxf(m_run, tmax);
    float acc = s_run * __expf(m_run - m_new);
#pragma unroll
    for (int r = 0; r < 8; ++r) acc += __expf(vals[r] - m_new);
    s_run = acc;
    m_run = m_new;
  }
  float m_o = __shfl_xor(m_run, 16, 32);
  float s_o = __shfl_xor(s_run, 16, 32);
  float m_f = fmaxf(m_run, m_o);
  float s_f = s_run * __expf(m_run - m_f) + s_o * __expf(m_o - m_f);
  if (lane < 16) {
    cm[(size_t)bh * T_SZ + s] = m_f;
    cs[(size_t)bh * T_SZ + s] = s_f;
  }
}

// ---------- 5) attention output: o = softmax_q(scores) @ v ----------
__global__ void attn_out_kernel(const h16* q, const h16* k, const h16* vT,
                                const float* cm, const float* cs, h16* o) {
  __shared__ h16 wt[8][16 * 32];  // per-wave 16x32 transpose staging
  int lane = threadIdx.x & 31;
  int wave = threadIdx.x >> 5;
  int bh = blockIdx.y;
  int b = bh / H_SZ;
  int head = bh % H_SZ;
  int t0 = (blockIdx.x * 8 + wave) * 16;
  const h16* qb = q + (size_t)bh * T_SZ * DH;
  const h16* kb = k + (size_t)bh * T_SZ * DH;
  const h16* vb = vT + (size_t)bh * (size_t)DH * T_SZ;
  const float* cmb = cm + (size_t)bh * T_SZ;
  const float* csb = cs + (size_t)bh * T_SZ;
  v16h a0 = frag_a(qb, t0, 0, DH, lane);
  v16h a1 = frag_a(qb, t0, 32, DH, lane);
  v8f oacc[4] = {{}, {}, {}, {}};
  int mrow = ((lane >> 4) << 3);
  for (int s0 = 0; s0 <= t0 + 15; s0 += 32) {
#pragma unroll
    for (int st = 0; st < 2; ++st) {
      int sb = s0 + st * 16;
      if (sb <= t0 + 15) {
        v8f c = {};
        c = wmma_f16(a0, frag_b(kb, sb, 0, DH, lane), c);
        c = wmma_f16(a1, frag_b(kb, sb, 32, DH, lane), c);
        int s = sb + (lane & 15);
        float cmv = cmb[s];
        float rcs = 1.0f / csb[s];
#pragma unroll
        for (int r = 0; r < 8; ++r) {
          int t = t0 + mrow + r;
          float w = (t >= s) ? __expf(8.0f * c[r] - cmv) * rcs : 0.0f;
          wt[wave][(mrow + r) * 32 + st * 16 + (lane & 15)] = (h16)w;
        }
      } else {
#pragma unroll
        for (int r = 0; r < 8; ++r)
          wt[wave][(mrow + r) * 32 + st * 16 + (lane & 15)] = (h16)0.0f;
      }
    }
    asm volatile("s_wait_dscnt 0" ::: "memory");
    v16h aw = frag_a(&wt[wave][0], 0, 0, 32, lane);
#pragma unroll
    for (int nt = 0; nt < 4; ++nt)
      oacc[nt] = wmma_f16(aw, frag_b(vb, nt * 16, s0, T_SZ, lane), oacc[nt]);
  }
  h16* ob = o + ((size_t)(b * T_SZ + t0)) * E_SZ + head * DH;
#pragma unroll
  for (int nt = 0; nt < 4; ++nt)
    for (int r = 0; r < 8; ++r)
      ob[(size_t)(mrow + r) * E_SZ + nt * 16 + (lane & 15)] = (h16)oacc[nt][r];
}

// ---------- 6) logits = o @ Wout + bout ----------
// Block: 256 threads = 8 waves in 2(M) x 4(N) layout.
// Block tile: 64M x 256N. Wave tile: 32M x 64N (2 A-frags x 4 B-frags = 8 WMMA/K).
// A panel (64x32 f16 = 4KB) staged in LDS via global_load_async_to_lds_b128 and
// shared across the 4 N-waves; B fragments double-buffered in registers.
__global__ void logits_kernel(const h16* o, const h16* WoutT, const float* bout,
                              float* out) {
  __shared__ h16 atile[64 * 32];  // 4 KB
  int tid = threadIdx.x;
  int lane = tid & 31;
  int wave = tid >> 5;
  int waveM = wave >> 2, waveN = wave & 3;
  int mblk = blockIdx.y * 64;
  int n0 = blockIdx.x * 256 + waveN * 64;

  // async-copy assignment: thread copies 16B; row = tid/4, chunk = (tid%4)*8 halfs
  int arow = tid >> 2;
  int achunk = (tid & 3) << 3;
  unsigned lds_dst =
      (unsigned)(uintptr_t)(void*)atile + (unsigned)(arow * 32 + achunk) * 2;
  const h16* gsrc_row = o + (size_t)(mblk + arow) * E_SZ + achunk;

  v8f acc[2][4] = {{{}, {}, {}, {}}, {{}, {}, {}, {}}};

  // columns (clamped for the V tail)
  int col[4], colc[4];
#pragma unroll
  for (int nt = 0; nt < 4; ++nt) {
    col[nt] = n0 + nt * 16 + (lane & 15);
    colc[nt] = col[nt] < V_SZ ? col[nt] : V_SZ - 1;
  }
  int koff_b = ((lane >> 4) << 4);

  // ---- prologue: stage first A panel + first B fragments ----
  async_copy_b128(lds_dst, gsrc_row);  // k0 = 0
  v16h bcur[4];
#pragma unroll
  for (int nt = 0; nt < 4; ++nt)
    bcur[nt] = *(const v16h*)(WoutT + (size_t)colc[nt] * E_SZ + koff_b);
  wait_async0();
  __syncthreads();

  for (int k0 = 0; k0 < E_SZ; k0 += 32) {
    // A fragments from LDS (shared across N-waves)
    v16h a0 = frag_a(atile, waveM * 32, 0, 32, lane);
    v16h a1 = frag_a(atile, waveM * 32 + 16, 0, 32, lane);
    __syncthreads();  // all waves done reading atile (dscnt waited before barrier)

    int kn = k0 + 32;
    v16h bnext[4];
    if (kn < E_SZ) {
      async_copy_b128(lds_dst, gsrc_row + kn);  // next A panel -> LDS
#pragma unroll
      for (int nt = 0; nt < 4; ++nt)
        bnext[nt] = *(const v16h*)(WoutT + (size_t)colc[nt] * E_SZ + kn + koff_b);
    }

#pragma unroll
    for (int nt = 0; nt < 4; ++nt) {
      acc[0][nt] = wmma_f16(a0, bcur[nt], acc[0][nt]);
      acc[1][nt] = wmma_f16(a1, bcur[nt], acc[1][nt]);
    }

    if (kn < E_SZ) {
      wait_async0();
#pragma unroll
      for (int nt = 0; nt < 4; ++nt) bcur[nt] = bnext[nt];
    }
    __syncthreads();  // next A panel visible to all waves
  }

  int mrow = ((lane >> 4) << 3);
#pragma unroll
  for (int mi = 0; mi < 2; ++mi) {
    int m0 = mblk + waveM * 32 + mi * 16;
#pragma unroll
    for (int nt = 0; nt < 4; ++nt) {
      if (col[nt] < V_SZ) {
        float bo = bout[col[nt]];
        for (int r = 0; r < 8; ++r)
          out[(size_t)(m0 + mrow + r) * V_SZ + col[nt]] = acc[mi][nt][r] + bo;
      }
    }
  }
}

// ---------- 7) per-row logsumexp + NLL ----------
__global__ void rowstats_kernel(const float* logits, const int* y, float* rowloss) {
  int m = blockIdx.x;
  int tid = threadIdx.x;
  const float* row = logits + (size_t)m * V_SZ;
  __shared__ float red[256];
  float mx = -3.0e38f;
  for (int v = tid; v < V_SZ; v += 256) mx = fmaxf(mx, row[v]);
  red[tid] = mx;
  __syncthreads();
  for (int st = 128; st > 0; st >>= 1) {
    if (tid < st) red[tid] = fmaxf(red[tid], red[tid + st]);
    __syncthreads();
  }
  float rowmax = red[0];
  __syncthreads();
  float sm = 0.f;
  for (int v = tid; v < V_SZ; v += 256) sm += __expf(row[v] - rowmax);
  red[tid] = sm;
  __syncthreads();
  for (int st = 128; st > 0; st >>= 1) {
    if (tid < st) red[tid] += red[tid + st];
    __syncthreads();
  }
  if (tid == 0) {
    float lse = rowmax + logf(red[0]);
    rowloss[m] = lse - row[y[m]];
  }
}

__global__ void loss_kernel(const float* rowloss, float* out_loss) {
  __shared__ float red[256];
  int tid = threadIdx.x;
  float s = 0.f;
  for (int i = tid; i < B_SZ * T_SZ; i += 256) s += rowloss[i];
  red[tid] = s;
  __syncthreads();
  for (int st = 128; st > 0; st >>= 1) {
    if (tid < st) red[tid] += red[tid + st];
    __syncthreads();
  }
  if (tid == 0) out_loss[0] = red[0] / (float)(B_SZ * T_SZ);
}

// ---------- host ----------
extern "C" void kernel_launch(void* const* d_in, const int* in_sizes, int n_in,
                              void* d_out, int out_size, void* d_ws, size_t ws_size,
                              hipStream_t stream) {
  (void)in_sizes; (void)n_in; (void)out_size; (void)ws_size;
  const int* x = (const int*)d_in[0];
  const int* y = (const int*)d_in[1];
  const float* tok = (const float*)d_in[2];
  const float* pos = (const float*)d_in[3];
  const float* Wq = (const float*)d_in[4];
  const float* Wk = (const float*)d_in[5];
  const float* Wv = (const float*)d_in[6];
  const float* Wout = (const float*)d_in[7];
  const float* bout = (const float*)d_in[8];
  float* out = (float*)d_out;

  char* ws = (char*)d_ws;
  size_t off = 0;
  auto take = [&](size_t bytes) -> char* {
    char* p = ws + off;
    off = (off + bytes + 255) & ~(size_t)255;
    return p;
  };
  h16* hbuf   = (h16*)take((size_t)B_SZ * T_SZ * E_SZ * 2);
  h16* Wqt    = (h16*)take((size_t)H_SZ * DH * E_SZ * 2);
  h16* Wkt    = (h16*)take((size_t)H_SZ * DH * E_SZ * 2);
  h16* Wvt    = (h16*)take((size_t)H_SZ * DH * E_SZ * 2);
  h16* WoutT  = (h16*)take((size_t)V_SZ * E_SZ * 2);
  h16* qb     = (h16*)take((size_t)B_SZ * H_SZ * T_SZ * DH * 2);
  h16* kb     = (h16*)take((size_t)B_SZ * H_SZ * T_SZ * DH * 2);
  h16* vTb    = (h16*)take((size_t)B_SZ * H_SZ * T_SZ * DH * 2);
  float* cm   = (float*)take((size_t)B_SZ * H_SZ * T_SZ * 4);
  float* cs   = (float*)take((size_t)B_SZ * H_SZ * T_SZ * 4);
  h16* obuf   = (h16*)take((size_t)B_SZ * T_SZ * E_SZ * 2);
  float* rl   = (float*)take((size_t)B_SZ * T_SZ * 4);
  float* loss = out + (size_t)B_SZ * T_SZ * V_SZ;

  embed_kernel<<<(B_SZ * T_SZ * E_SZ) / 256, 256, 0, stream>>>(x, tok, pos, hbuf);
  prep_w_kernel<<<(H_SZ * E_SZ * DH) / 256, 256, 0, stream>>>(Wq, Wk, Wv, Wqt, Wkt,
                                                              Wvt);
  prep_wout_kernel<<<(int)(((size_t)E_SZ * V_SZ + 255) / 256), 256, 0, stream>>>(
      Wout, WoutT);
  qkv_kernel<<<dim3(T_SZ / 128, B_SZ * H_SZ), 256, 0, stream>>>(hbuf, Wqt, Wkt, Wvt,
                                                                qb, kb, vTb);
  colstats_kernel<<<dim3(T_SZ / 64, B_SZ * H_SZ), 128, 0, stream>>>(qb, kb, cm, cs);
  attn_out_kernel<<<dim3(T_SZ / 128, B_SZ * H_SZ), 256, 0, stream>>>(qb, kb, vTb, cm,
                                                                     cs, obuf);
  logits_kernel<<<dim3((V_SZ + 255) / 256, (B_SZ * T_SZ) / 64), 256, 0, stream>>>(
      obuf, WoutT, bout, out);
  rowstats_kernel<<<B_SZ * T_SZ, 256, 0, stream>>>(out, y, rl);
  loss_kernel<<<1, 256, 0, stream>>>(rl, loss);
}
